// GNN_LSTM_16226386444613
// MI455X (gfx1250) — matrix operand order repaired and checked by reading, not compiled, not run on hardware
//
#include <hip/hip_runtime.h>
#include <math.h>

// ---------------- problem constants ----------------
constexpr int GN   = 10000;   // nodes
constexpr int GF   = 64;      // features
constexpr int GT   = 8;       // timesteps
constexpr int GE   = 160000;  // edges
constexpr int GTS  = 200;     // time-series length
constexpr int GH   = 128;     // hidden
constexpr int G4H  = 512;     // 4*H
constexpr int GKC  = 192;     // F + H (fused cell GEMM K)
constexpr int GK2  = 384;     // 3*H (map GEMM K)
constexpr int GTSM = 208;     // TS padded to mult of 16
constexpr int GKTS = 10016;   // N padded to mult of 32
constexpr int GKTOP= 5000;    // K of top-k
constexpr int GS   = 16384;   // sort size (pow2 >= N)
constexpr float GEPS = 1e-8f;

// ---------------- WMMA types ----------------
typedef __attribute__((ext_vector_type(16))) __bf16 v16bf;
typedef __attribute__((ext_vector_type(8)))  __bf16 v8bf;
typedef __attribute__((ext_vector_type(8)))  float  v8f;

__device__ __forceinline__ v16bf gl_cat8(v8bf a, v8bf b) {
  return __builtin_shufflevector(a, b, 0,1,2,3,4,5,6,7,8,9,10,11,12,13,14,15);
}
__device__ __forceinline__ unsigned short gl_f2bf(float f) {
  unsigned u = __builtin_bit_cast(unsigned, f);
  u += 0x7FFFu + ((u >> 16) & 1u);            // round-to-nearest-even
  return (unsigned short)(u >> 16);
}
__device__ __forceinline__ float gl_sigmoid(float x) { return 1.0f / (1.0f + expf(-x)); }

// ---------------- generic helpers ----------------
__global__ void gl_fill(float* p, float v, int n) {
  int i = blockIdx.x * blockDim.x + threadIdx.x;
  if (i < n) p[i] = v;
}
__global__ void gl_copy(float* d, const float* s, int n) {
  int i = blockIdx.x * blockDim.x + threadIdx.x;
  if (i < n) d[i] = s[i];
}

// ---------------- WMMA GEMM: C(MxNout) = A(MxK) @ B(NoutxK)^T, bf16 in f32 out
// grid: (M/16, Nout/128), block: 64 threads (2 waves); wave tile = 16 x 64.
// Software-pipelined: iteration k+1's fragments are loaded into a rotating
// register set before iteration k's WMMAs execute, so VMEM latency overlaps
// the matrix pipe instead of serializing on s_wait_loadcnt 0.
__global__ void gl_wmma_gemm(const unsigned short* __restrict__ A,
                             const unsigned short* __restrict__ B,
                             float* __restrict__ C, int K, int Nout) {
  const int wave = threadIdx.x >> 5;
  const int lane = threadIdx.x & 31;
  const int l  = lane & 15;
  const int hi = lane >> 4;
  const int m0 = blockIdx.x * 16;
  const int colBase = (blockIdx.y * 2 + wave) * 64;

  v8f acc[4] = {};
  const unsigned short* arow = A + (size_t)(m0 + l) * K;
  const unsigned short* brow[4];
#pragma unroll
  for (int nt = 0; nt < 4; ++nt)
    brow[nt] = B + (size_t)(colBase + nt * 16 + l) * K;

  // A fragment: elems 0..7 -> K = k0+hi*8+e ; elems 8..15 -> K = k0+16+hi*8+e
  // B fragment (col n = colBase+nt*16+l): same K mapping as A fragment
  v8bf alo, ahi, blo[4], bhi[4];
  alo = *reinterpret_cast<const v8bf*>(arow + hi * 8);
  ahi = *reinterpret_cast<const v8bf*>(arow + 16 + hi * 8);
#pragma unroll
  for (int nt = 0; nt < 4; ++nt) {
    blo[nt] = *reinterpret_cast<const v8bf*>(brow[nt] + hi * 8);
    bhi[nt] = *reinterpret_cast<const v8bf*>(brow[nt] + 16 + hi * 8);
  }

  for (int k0 = 32; k0 < K; k0 += 32) {
    // ---- prefetch next iteration's fragments (no dependence on WMMAs below)
    v8bf nalo = *reinterpret_cast<const v8bf*>(arow + k0 + hi * 8);
    v8bf nahi = *reinterpret_cast<const v8bf*>(arow + k0 + 16 + hi * 8);
    v8bf nblo[4], nbhi[4];
#pragma unroll
    for (int nt = 0; nt < 4; ++nt) {
      nblo[nt] = *reinterpret_cast<const v8bf*>(brow[nt] + k0 + hi * 8);
      nbhi[nt] = *reinterpret_cast<const v8bf*>(brow[nt] + k0 + 16 + hi * 8);
    }
    if (k0 + 32 < K) __builtin_prefetch(arow + k0 + 32, 0, 3);  // near-scope prefetch
    // ---- consume current fragments with the matrix pipe
    v16bf af = gl_cat8(alo, ahi);
#pragma unroll
    for (int nt = 0; nt < 4; ++nt) {
      v16bf bf = gl_cat8(blo[nt], bhi[nt]);
      acc[nt] = __builtin_amdgcn_wmma_f32_16x16x32_bf16(
          false, af, false, bf, (short)0, acc[nt], false, false);
    }
    // ---- rotate
    alo = nalo; ahi = nahi;
#pragma unroll
    for (int nt = 0; nt < 4; ++nt) { blo[nt] = nblo[nt]; bhi[nt] = nbhi[nt]; }
  }
  // ---- epilogue: last k-step
  {
    v16bf af = gl_cat8(alo, ahi);
#pragma unroll
    for (int nt = 0; nt < 4; ++nt) {
      v16bf bf = gl_cat8(blo[nt], bhi[nt]);
      acc[nt] = __builtin_amdgcn_wmma_f32_16x16x32_bf16(
          false, af, false, bf, (short)0, acc[nt], false, false);
    }
  }
#pragma unroll
  for (int nt = 0; nt < 4; ++nt) {
    int col = colBase + nt * 16 + l;
#pragma unroll
    for (int v = 0; v < 8; ++v)
      C[(size_t)(m0 + hi * 8 + v) * Nout + col] = acc[nt][v];
  }
}

// ---------------- graph prep ----------------
__global__ void gl_deg_edges(const int* __restrict__ dst, float* deg) {
  int e = blockIdx.x * blockDim.x + threadIdx.x;
  if (e < GE) atomicAdd(&deg[dst[e]], 1.0f);
}
__global__ void gl_deg_fin(const float* deg, float* dis, float* self_n) {
  int i = blockIdx.x * blockDim.x + threadIdx.x;
  if (i < GN) { float r = rsqrtf(deg[i]); dis[i] = r; self_n[i] = r * r; }
}
__global__ void gl_enorm(const int* __restrict__ src, const int* __restrict__ dst,
                         const float* dis, float* enorm) {
  int e = blockIdx.x * blockDim.x + threadIdx.x;
  if (e < GE) enorm[e] = dis[src[e]] * dis[dst[e]];
}

// ---------------- weight / input packing (fp32 -> bf16) ----------------
__global__ void gl_build_wc(const float* __restrict__ Wx, const float* __restrict__ Wh,
                            unsigned short* Wc) {
  int idx = blockIdx.x * blockDim.x + threadIdx.x;
  if (idx >= G4H * GKC) return;
  int row = idx / GKC, k = idx % GKC;             // row = g*H + j
  float v = (k < GF) ? Wx[(size_t)row * GF + k] : Wh[(size_t)row * GH + (k - GF)];
  Wc[idx] = gl_f2bf(v);
}
__global__ void gl_build_bias(const float* bx, const float* bh, float* biasC) {
  int i = blockIdx.x * blockDim.x + threadIdx.x;
  if (i < G4H) biasC[i] = bx[i] + bh[i];
}
__global__ void gl_pack_z(const float* __restrict__ xt, const float* __restrict__ h,
                          unsigned short* Z) {
  int idx = blockIdx.x * blockDim.x + threadIdx.x;
  if (idx >= GN * GKC) return;
  int n = idx / GKC, k = idx % GKC;
  float v = (k < GF) ? xt[(size_t)n * GF + k] : h[(size_t)n * GH + (k - GF)];
  Z[idx] = gl_f2bf(v);
}
__global__ void gl_build_z2(const float* __restrict__ l0, const float* __restrict__ l1,
                            const float* __restrict__ l2, unsigned short* Z2) {
  int idx = blockIdx.x * blockDim.x + threadIdx.x;
  if (idx >= GN * GK2) return;
  int n = idx / GK2, k = idx % GK2;
  int j = k / GH, d = k % GH;
  const float* p = (j == 0) ? l0 : (j == 1) ? l1 : l2;
  Z2[idx] = gl_f2bf(p[(size_t)n * GH + d]);
}
__global__ void gl_build_wm(const float* __restrict__ mapW, unsigned short* Wm) {
  int idx = blockIdx.x * blockDim.x + threadIdx.x;
  if (idx >= GH * GK2) return;
  int h = idx / GK2, k = idx % GK2;
  int j = k / GH, d = k % GH;
  Wm[idx] = gl_f2bf(mapW[((size_t)j * GH + h) * GH + d]);
}
__global__ void gl_pack_ts(const float* __restrict__ ts, unsigned short* TSa) {
  int idx = blockIdx.x * blockDim.x + threadIdx.x;
  if (idx >= GTSM * GKTS) return;
  int t = idx / GKTS, n = idx % GKTS;
  float v = (t < GTS && n < GN) ? ts[(size_t)t * GN + n] : 0.0f;
  TSa[idx] = gl_f2bf(v);
}
__global__ void gl_pack_wih(const float* __restrict__ Wih, unsigned short* Wihb) {
  int idx = blockIdx.x * blockDim.x + threadIdx.x;
  if (idx >= G4H * GKTS) return;
  int r = idx / GKTS, k = idx % GKTS;
  Wihb[idx] = gl_f2bf((k < GN) ? Wih[(size_t)r * GN + k] : 0.0f);
}

// ---------------- aggregation + cell gates ----------------
__global__ void gl_agg_init(const float* __restrict__ lin, const float* __restrict__ self_n,
                            const float* __restrict__ biasC, float* agg) {
  int idx = blockIdx.x * blockDim.x + threadIdx.x;
  if (idx >= GN * G4H) return;
  int n = idx / G4H, m = idx % G4H;
  agg[idx] = lin[idx] * self_n[n] + biasC[m];
}
__global__ void gl_agg_edges(const float* __restrict__ lin, const float* __restrict__ enorm,
                             const int* __restrict__ src, const int* __restrict__ dst,
                             float* agg) {
  int e = blockIdx.x;
  int s = src[e], d = dst[e];
  float w = enorm[e];
  const float* ls = lin + (size_t)s * G4H;
  float* ad = agg + (size_t)d * G4H;
  for (int j = threadIdx.x; j < G4H; j += blockDim.x)
    atomicAdd(&ad[j], ls[j] * w);
}
__global__ void gl_gates(const float* __restrict__ agg, float* c, float* h,
                         const float* __restrict__ lng, const float* __restrict__ lnb) {
  int n = blockIdx.x, j = threadIdx.x;                // 128 threads
  const float* p = agg + (size_t)n * G4H;
  float pi = p[j], pf = p[GH + j], po = p[2 * GH + j], pm = p[3 * GH + j];
  __shared__ float red[GH];
  red[j] = pm; __syncthreads();
  for (int s = 64; s > 0; s >>= 1) { if (j < s) red[j] += red[j + s]; __syncthreads(); }
  float mean = red[0] * (1.0f / GH);
  __syncthreads();
  float dv = pm - mean;
  red[j] = dv * dv; __syncthreads();
  for (int s = 64; s > 0; s >>= 1) { if (j < s) red[j] += red[j + s]; __syncthreads(); }
  float var = red[0] * (1.0f / GH);
  float g  = tanhf(dv * rsqrtf(var + 1e-5f) * lng[j] + lnb[j]);
  float i_ = fmaxf(pi, 0.0f), f_ = fmaxf(pf, 0.0f), o_ = fmaxf(po, 0.0f);
  size_t off = (size_t)n * GH + j;
  float cn = f_ * c[off] + i_ * g;
  c[off] = cn;
  h[off] = o_ * tanhf(cn);
}

// ---------------- DGPool ----------------
__global__ void gl_pool_norm(const float* v, float* vn) {
  int j = threadIdx.x;                               // 128
  __shared__ float red[GH];
  red[j] = v[j] * v[j]; __syncthreads();
  for (int s = 64; s > 0; s >>= 1) { if (j < s) red[j] += red[j + s]; __syncthreads(); }
  float nrm = sqrtf(red[0]);
  vn[j] = v[j] / (nrm + GEPS);
}
__global__ void gl_score(const float* __restrict__ hidden, const float* __restrict__ vn,
                         float* sc) {
  int n = blockIdx.x, j = threadIdx.x;               // 128
  __shared__ float red[GH];
  red[j] = hidden[(size_t)n * GH + j] * vn[j]; __syncthreads();
  for (int s = 64; s > 0; s >>= 1) { if (j < s) red[j] += red[j + s]; __syncthreads(); }
  if (j == 0) sc[n] = red[0];
}
__global__ void gl_stats_reduce(const float* sc, float* stats) {
  __shared__ float s1[256], s2[256];
  float a = 0, b = 0;
  for (int i = blockIdx.x * blockDim.x + threadIdx.x; i < GN; i += gridDim.x * blockDim.x) {
    float x = sc[i]; a += x; b += x * x;
  }
  s1[threadIdx.x] = a; s2[threadIdx.x] = b; __syncthreads();
  for (int s = 128; s > 0; s >>= 1) {
    if (threadIdx.x < s) { s1[threadIdx.x] += s1[threadIdx.x + s]; s2[threadIdx.x] += s2[threadIdx.x + s]; }
    __syncthreads();
  }
  if (threadIdx.x == 0) { atomicAdd(&stats[0], s1[0]); atomicAdd(&stats[1], s2[0]); }
}
__global__ void gl_stats_fin(float* stats) {
  float mean = stats[0] / GN;
  float var  = stats[1] / GN - mean * mean;
  stats[2] = mean;
  stats[3] = sqrtf(fmaxf(var, 0.0f));
}
__global__ void gl_sigmoid_k(const float* sc, const float* stats, float* sig) {
  int i = blockIdx.x * blockDim.x + threadIdx.x;
  if (i < GN) sig[i] = gl_sigmoid((sc[i] - stats[2]) / (stats[3] + GEPS));
}
// single-block bitonic sort (descending) of 16384 (value,index) pairs in dynamic LDS
__global__ void gl_bitonic(const float* __restrict__ sig, float* skeys, int* sidx) {
  extern __shared__ unsigned char smem[];
  float* k = (float*)smem;
  int*   v = (int*)(smem + (size_t)GS * sizeof(float));
  for (int i = threadIdx.x; i < GS; i += blockDim.x) {
    bool in = i < GN;
    k[i] = in ? sig[i] : -1.0e30f;
    v[i] = in ? i : 0;
  }
  __syncthreads();
  for (int len = 2; len <= GS; len <<= 1) {
    for (int st = len >> 1; st > 0; st >>= 1) {
      for (int j = threadIdx.x; j < GS; j += blockDim.x) {
        int x = j ^ st;
        if (x > j) {
          bool desc = ((j & len) == 0);
          float kj = k[j], kx = k[x];
          if (desc ? (kj < kx) : (kj > kx)) {
            k[j] = kx; k[x] = kj;
            int t = v[j]; v[j] = v[x]; v[x] = t;
          }
        }
      }
      __syncthreads();
    }
  }
  for (int i = threadIdx.x; i < GS; i += blockDim.x) { skeys[i] = k[i]; sidx[i] = v[i]; }
}
__global__ void gl_loss_reduce(const float* skeys, float* lossA) {
  __shared__ float red[256];
  float a = 0;
  for (int r = blockIdx.x * blockDim.x + threadIdx.x; r < GN; r += gridDim.x * blockDim.x) {
    float ss = skeys[r];
    a += (r < GKTOP) ? logf(ss + GEPS) : logf(1.0f - ss + GEPS);
  }
  red[threadIdx.x] = a; __syncthreads();
  for (int s = 128; s > 0; s >>= 1) { if (threadIdx.x < s) red[threadIdx.x] += red[threadIdx.x + s]; __syncthreads(); }
  if (threadIdx.x == 0) atomicAdd(lossA, red[0]);
}
__global__ void gl_high(const float* __restrict__ hidden, const float* __restrict__ skeys,
                        const int* __restrict__ sidx, float* high) {
  int c = blockIdx.x;                                // 0..127
  __shared__ float red[256];
  float a = 0;
  for (int r = threadIdx.x; r < GKTOP; r += blockDim.x)
    a += skeys[r] * hidden[(size_t)sidx[r] * GH + c];
  red[threadIdx.x] = a; __syncthreads();
  for (int s = 128; s > 0; s >>= 1) { if (threadIdx.x < s) red[threadIdx.x] += red[threadIdx.x + s]; __syncthreads(); }
  if (threadIdx.x == 0) high[c] = red[0] / (float)GKTOP;
}

// ---------------- sequential LSTM scan (after hoisting Wih@x GEMM) ----------------
__global__ void gl_lstm_scan(const float* __restrict__ U, const float* __restrict__ Whh,
                             const float* __restrict__ bih, const float* __restrict__ bhh,
                             float* hlast) {
  int tid = threadIdx.x;                             // 512
  __shared__ float hs[GH], cs[GH], zs[G4H];
  if (tid < GH) { hs[tid] = 0.0f; cs[tid] = 0.0f; }
  float bsum = bih[tid] + bhh[tid];
  __syncthreads();
  for (int t = 0; t < GTS; ++t) {
    float acc = U[(size_t)t * G4H + tid] + bsum;
    const float* wr = Whh + (size_t)tid * GH;
#pragma unroll 4
    for (int kk = 0; kk < GH; ++kk) acc += wr[kk] * hs[kk];
    zs[tid] = acc;
    __syncthreads();
    if (tid < GH) {
      float i_ = gl_sigmoid(zs[tid]);
      float f_ = gl_sigmoid(zs[GH + tid]);
      float g_ = tanhf(zs[2 * GH + tid]);
      float o_ = gl_sigmoid(zs[3 * GH + tid]);
      float cn = f_ * cs[tid] + i_ * g_;
      cs[tid] = cn;
      hs[tid] = o_ * tanhf(cn);
    }
    __syncthreads();
  }
  if (tid < GH) hlast[tid] = hs[tid];
}

// ---------------- final MLP head ----------------
__global__ void gl_mlp(const float* __restrict__ high, const float* __restrict__ hlast,
                       const float* __restrict__ W1, const float* __restrict__ b1,
                       const float* __restrict__ lng, const float* __restrict__ lnb,
                       const float* __restrict__ W2, const float* __restrict__ b2,
                       const float* __restrict__ W3, const float* __restrict__ b3,
                       const float* __restrict__ lossA, float* out) {
  int tid = threadIdx.x;                             // 256
  __shared__ float fu[2 * GH], m1[GH], m1n[GH], m2[GH / 2], stat[2];
  fu[tid] = (tid < GH) ? high[tid] : hlast[tid - GH];
  __syncthreads();
  if (tid < GH) {
    float acc = b1[tid];
    const float* wr = W1 + (size_t)tid * (2 * GH);
    for (int kk = 0; kk < 2 * GH; ++kk) acc += wr[kk] * fu[kk];
    m1[tid] = fmaxf(acc, 0.0f);
  }
  __syncthreads();
  if (tid == 0) {
    float s = 0; for (int i = 0; i < GH; ++i) s += m1[i];
    float mean = s / GH;
    float v = 0; for (int i = 0; i < GH; ++i) { float d = m1[i] - mean; v += d * d; }
    stat[0] = mean; stat[1] = v / GH;
  }
  __syncthreads();
  if (tid < GH)
    m1n[tid] = (m1[tid] - stat[0]) * rsqrtf(stat[1] + 1e-5f) * lng[tid] + lnb[tid];
  __syncthreads();
  if (tid < GH / 2) {
    float acc = b2[tid];
    const float* wr = W2 + (size_t)tid * GH;
    for (int kk = 0; kk < GH; ++kk) acc += wr[kk] * m1n[kk];
    m2[tid] = fmaxf(acc, 0.0f);
  }
  __syncthreads();
  if (tid == 0) {
    float p = b3[0];
    for (int kk = 0; kk < GH / 2; ++kk) p += W3[kk] * m2[kk];
    out[0] = p;
    out[1] = -lossA[0] / (float)GN;
  }
}

// ---------------- driver ----------------
extern "C" void kernel_launch(void* const* d_in, const int* in_sizes, int n_in,
                              void* d_out, int out_size, void* d_ws, size_t ws_size,
                              hipStream_t stream) {
  (void)in_sizes; (void)n_in; (void)out_size; (void)ws_size;
  const float* x_seq  = (const float*)d_in[0];
  const float* h0     = (const float*)d_in[1];
  const float* c0     = (const float*)d_in[2];
  const float* ts     = (const float*)d_in[3];
  const float* Wx     = (const float*)d_in[4];
  const float* bx     = (const float*)d_in[5];
  const float* Wh     = (const float*)d_in[6];
  const float* bh     = (const float*)d_in[7];
  const float* lng    = (const float*)d_in[8];
  const float* lnb    = (const float*)d_in[9];
  const float* mapW   = (const float*)d_in[10];
  const float* pvec   = (const float*)d_in[11];
  const float* Wih    = (const float*)d_in[12];
  const float* Whh    = (const float*)d_in[13];
  const float* bih    = (const float*)d_in[14];
  const float* bhh    = (const float*)d_in[15];
  const float* W1     = (const float*)d_in[16];
  const float* b1     = (const float*)d_in[17];
  const float* mlng   = (const float*)d_in[18];
  const float* mlnb   = (const float*)d_in[19];
  const float* W2     = (const float*)d_in[20];
  const float* b2     = (const float*)d_in[21];
  const float* W3     = (const float*)d_in[22];
  const float* b3     = (const float*)d_in[23];
  const int*   ei     = (const int*)d_in[24];
  const int*   esrc   = ei;
  const int*   edst   = ei + GE;
  float* out = (float*)d_out;

  // ---- workspace carve ----
  unsigned char* w = (unsigned char*)d_ws;
  size_t off = 0;
  auto take = [&](size_t bytes) { size_t o = off; off += (bytes + 255) & ~(size_t)255; return o; };
  unsigned short* Wc    = (unsigned short*)(w + take((size_t)G4H * GKC * 2));
  unsigned short* Zb    = (unsigned short*)(w + take((size_t)GN * GKC * 2));
  float* lin   = (float*)(w + take((size_t)GN * G4H * 4));
  float* agg   = (float*)(w + take((size_t)GN * G4H * 4));
  float* cbuf  = (float*)(w + take((size_t)GN * GH * 4));
  float* last0 = (float*)(w + take((size_t)GN * GH * 4));
  float* last1 = (float*)(w + take((size_t)GN * GH * 4));
  float* last2 = (float*)(w + take((size_t)GN * GH * 4));
  float* deg   = (float*)(w + take((size_t)GN * 4));
  float* dis   = (float*)(w + take((size_t)GN * 4));
  float* selfn = (float*)(w + take((size_t)GN * 4));
  float* enorm = (float*)(w + take((size_t)GE * 4));
  float* biasC = (float*)(w + take((size_t)G4H * 4));
  unsigned short* Z2 = (unsigned short*)(w + take((size_t)GN * GK2 * 2));
  unsigned short* Wm = (unsigned short*)(w + take((size_t)GH * GK2 * 2));
  float* hidden = (float*)(w + take((size_t)GN * GH * 4));
  float* vnorm  = (float*)(w + take((size_t)GH * 4));
  float* sc     = (float*)(w + take((size_t)GN * 4));
  float* sig    = (float*)(w + take((size_t)GN * 4));
  float* stats  = (float*)(w + take(4 * 4));
  float* skeys  = (float*)(w + take((size_t)GS * 4));
  int*   sidx   = (int*)  (w + take((size_t)GS * 4));
  float* lossA  = (float*)(w + take(4));
  float* high   = (float*)(w + take((size_t)GH * 4));
  float* Ubuf   = (float*)(w + take((size_t)GTSM * G4H * 4));
  float* hlast  = (float*)(w + take((size_t)GH * 4));
  // big bf16 staging buffers alias the dead lin/agg regions (GNN phase is over)
  unsigned short* TSa  = (unsigned short*)lin;   // 208*10016*2 = 4.2 MB  <= 20.5 MB
  unsigned short* Wihb = (unsigned short*)agg;   // 512*10016*2 = 10.3 MB <= 20.5 MB

  auto cdiv = [](int a, int b) { return (a + b - 1) / b; };

  // ---- graph norm prep ----
  gl_fill<<<cdiv(GN, 256), 256, 0, stream>>>(deg, 1.0f, GN);          // self loop
  gl_deg_edges<<<cdiv(GE, 256), 256, 0, stream>>>(edst, deg);
  gl_deg_fin<<<cdiv(GN, 256), 256, 0, stream>>>(deg, dis, selfn);
  gl_enorm<<<cdiv(GE, 256), 256, 0, stream>>>(esrc, edst, dis, enorm);

  // ---- fused cell weights / bias ----
  gl_build_wc<<<cdiv(G4H * GKC, 256), 256, 0, stream>>>(Wx, Wh, Wc);
  gl_build_bias<<<cdiv(G4H, 256), 256, 0, stream>>>(bx, bh, biasC);

  // ---- three jump-connection branches ----
  struct Branch { int start, stride; float* hbuf; };
  Branch br[3] = { {0, 1, last0}, {0, 2, last1}, {1, 2, last2} };
  for (int b = 0; b < 3; ++b) {
    gl_copy<<<cdiv(GN * GH, 256), 256, 0, stream>>>(br[b].hbuf, h0, GN * GH);
    gl_copy<<<cdiv(GN * GH, 256), 256, 0, stream>>>(cbuf, c0, GN * GH);
    for (int t = br[b].start; t < GT; t += br[b].stride) {
      gl_pack_z<<<cdiv(GN * GKC, 256), 256, 0, stream>>>(
          x_seq + (size_t)t * GN * GF, br[b].hbuf, Zb);
      gl_wmma_gemm<<<dim3(GN / 16, G4H / 128), 64, 0, stream>>>(Zb, Wc, lin, GKC, G4H);
      gl_agg_init<<<cdiv(GN * G4H, 256), 256, 0, stream>>>(lin, selfn, biasC, agg);
      gl_agg_edges<<<GE, 128, 0, stream>>>(lin, enorm, esrc, edst, agg);
      gl_gates<<<GN, GH, 0, stream>>>(agg, cbuf, br[b].hbuf, lng, lnb);
    }
  }

  // ---- hidden = sum_j last_j @ map_W[j]^T  (fused K=384 WMMA GEMM) ----
  gl_build_z2<<<cdiv(GN * GK2, 256), 256, 0, stream>>>(last0, last1, last2, Z2);
  gl_build_wm<<<cdiv(GH * GK2, 256), 256, 0, stream>>>(mapW, Wm);
  gl_wmma_gemm<<<dim3(GN / 16, GH / 128), 64, 0, stream>>>(Z2, Wm, hidden, GK2, GH);

  // ---- DGPool ----
  gl_pool_norm<<<1, GH, 0, stream>>>(pvec, vnorm);
  gl_score<<<GN, GH, 0, stream>>>(hidden, vnorm, sc);
  gl_fill<<<1, 32, 0, stream>>>(stats, 0.0f, 4);
  gl_stats_reduce<<<128, 256, 0, stream>>>(sc, stats);
  gl_stats_fin<<<1, 1, 0, stream>>>(stats);
  gl_sigmoid_k<<<cdiv(GN, 256), 256, 0, stream>>>(sc, stats, sig);
  gl_bitonic<<<1, 1024, (size_t)GS * 8, stream>>>(sig, skeys, sidx);
  gl_fill<<<1, 32, 0, stream>>>(lossA, 0.0f, 1);
  gl_loss_reduce<<<40, 256, 0, stream>>>(skeys, lossA);
  gl_high<<<GH, 256, 0, stream>>>(hidden, skeys, sidx, high);

  // ---- time-series LSTM: hoisted Wih GEMM + sequential scan ----
  gl_pack_ts<<<cdiv(GTSM * GKTS, 256), 256, 0, stream>>>(ts, TSa);
  gl_pack_wih<<<cdiv(G4H * GKTS, 256), 256, 0, stream>>>(Wih, Wihb);
  gl_wmma_gemm<<<dim3(GTSM / 16, G4H / 128), 64, 0, stream>>>(TSa, Wihb, Ubuf, GKTS, G4H);
  gl_lstm_scan<<<1, G4H, 0, stream>>>(Ubuf, Whh, bih, bhh, hlast);

  // ---- MLP head + outputs ----
  gl_mlp<<<1, 256, 0, stream>>>(high, hlast, W1, b1, mlng, mlnb, W2, b2, W3, b3, lossA, out);
}
